// StudentModel_49924699848778
// MI455X (gfx1250) — matrix operand order repaired and checked by reading, not compiled, not run on hardware
//
#include <hip/hip_runtime.h>

// GAT x3 for MI455X (gfx1250, wave32). Dense GEMMs in bf16 WMMA (fp32 acc),
// edge softmax/scatter via L2 float atomics (all hot buffers fit in 192MB L2).

#define NNODES 20000
#define NEG_SLOPE 0.2f
#define LN_EPS 1e-5f

typedef __attribute__((ext_vector_type(16))) __bf16 v16bf;
typedef __attribute__((ext_vector_type(8)))  __bf16 v8bf;
typedef __attribute__((ext_vector_type(8)))  float  v8f;

// ---------------- conversion kernels ----------------

// fp32 [M,Kin] -> bf16 [M,Kpad] (zero pad K)
__global__ void cvt_pad_bf16_kernel(const float* __restrict__ src, __bf16* __restrict__ dst,
                                    int M, int Kin, int Kpad) {
  long i = (long)blockIdx.x * blockDim.x + threadIdx.x;
  long total = (long)M * Kpad;
  if (i >= total) return;
  int  k = (int)(i % Kpad);
  long m = i / Kpad;
  dst[i] = (k < Kin) ? (__bf16)src[m * Kin + k] : (__bf16)0.0f;
}

// fp32 W[Kin,Nw] -> bf16 Wt[Nw,Kpad] transposed, zero pad K
__global__ void wtrans_bf16_kernel(const float* __restrict__ W, __bf16* __restrict__ Wt,
                                   int Kin, int Nw, int Kpad) {
  long i = (long)blockIdx.x * blockDim.x + threadIdx.x;
  long total = (long)Nw * Kpad;
  if (i >= total) return;
  int  k = (int)(i % Kpad);
  long n = i / Kpad;
  Wt[i] = (k < Kin) ? (__bf16)W[(long)k * Nw + n] : (__bf16)0.0f;
}

__global__ void zero_f32_kernel(float* __restrict__ p, long n) {
  long i = (long)blockIdx.x * blockDim.x + threadIdx.x;
  if (i < n) p[i] = 0.0f;
}

__global__ void init_softmax_kernel(unsigned* __restrict__ mmax, float* __restrict__ denom, int n) {
  int i = blockIdx.x * blockDim.x + threadIdx.x;
  if (i < n) { mmax[i] = 0u; denom[i] = 0.0f; }
}

// ---------------- WMMA GEMM: C[M,N] = A[M,K](bf16) * Bt[N,K]^T(bf16) (+bias) ----------------
// Block: 256 thr = 8 waves; tile 128(M)x128(N); wave = 32x64 = 2x4 WMMA 16x16 tiles.
// Fragment layouts per CDNA5 ISA 7.12.2 (wave32):
//   A 16x32 bf16 : row m = lane&15 ; K = e + ((e>=8)?8:0) + 8*(lane>>4)  -> two 16B loads
//   B 32x16 bf16 : col n = lane&15 ; K = e + ((e>=8)?8:0) + 8*(lane>>4)                -> two 16B loads (N-major Bt)
//   C/D 16x16 f32: N = lane&15 ; M = r + 8*(lane>>4)
__global__ __launch_bounds__(256)
void gemm_bf16_wmma_kernel(const __bf16* __restrict__ A,
                           const __bf16* __restrict__ Bt,
                           float* __restrict__ C,
                           int M, int Nn, int K, int ldc,
                           const float* __restrict__ bias1,
                           const float* __restrict__ bias2) {
  const int lane = threadIdx.x & 31;
  const int wave = threadIdx.x >> 5;
  const int ll   = lane & 15;
  const int lh   = lane >> 4;
  const int bm   = blockIdx.x * 128 + (wave & 3) * 32;
  const int bn   = blockIdx.y * 128 + (wave >> 2) * 64;

  v8f acc[2][4];
#pragma unroll
  for (int mi = 0; mi < 2; mi++)
#pragma unroll
    for (int ni = 0; ni < 4; ni++)
#pragma unroll
      for (int e = 0; e < 8; e++) acc[mi][ni][e] = 0.0f;

  long arow[2];
#pragma unroll
  for (int mi = 0; mi < 2; mi++) {
    int m = bm + mi * 16 + ll;
    if (m > M - 1) m = M - 1;           // clamp; OOB rows never stored
    arow[mi] = (long)m * K;
  }
  int ncol[4]; long brow[4];
#pragma unroll
  for (int ni = 0; ni < 4; ni++) {
    int n = bn + ni * 16 + ll;
    ncol[ni] = n;
    brow[ni] = (long)((n < Nn) ? n : 0) * K; // clamp; OOB cols never stored
  }

  for (int kb = 0; kb < K; kb += 32) {
    const int ka = kb + lh * 8;
    v16bf af[2];
#pragma unroll
    for (int mi = 0; mi < 2; mi++) {
      v8bf lo = *(const v8bf*)(A + arow[mi] + ka);
      v8bf hi = *(const v8bf*)(A + arow[mi] + ka + 16);
      af[mi] = __builtin_shufflevector(lo, hi, 0,1,2,3,4,5,6,7,8,9,10,11,12,13,14,15);
    }
    const int kc = kb + lh * 8;
    v16bf bfv[4];
#pragma unroll
    for (int ni = 0; ni < 4; ni++) {
      v8bf lo = *(const v8bf*)(Bt + brow[ni] + kc);
      v8bf hi = *(const v8bf*)(Bt + brow[ni] + kc + 16);
      bfv[ni] = __builtin_shufflevector(lo, hi, 0,1,2,3,4,5,6,7,8,9,10,11,12,13,14,15);
    }
    if (kb + 32 < K) {  // pull next K-slab toward L1 (global_prefetch_b8)
      __builtin_prefetch(A  + arow[0] + ka + 32, 0, 1);
      __builtin_prefetch(Bt + brow[0] + kc + 32, 0, 1);
    }
#pragma unroll
    for (int mi = 0; mi < 2; mi++)
#pragma unroll
      for (int ni = 0; ni < 4; ni++)
        acc[mi][ni] = __builtin_amdgcn_wmma_f32_16x16x32_bf16(
            false, af[mi], false, bfv[ni], (short)0, acc[mi][ni], false, false);
  }

#pragma unroll
  for (int ni = 0; ni < 4; ni++) {
    const int nc = ncol[ni];
    if (nc >= Nn) continue;
    float bb = 0.0f;
    if (bias1) bb += bias1[nc];
    if (bias2) bb += bias2[nc];
#pragma unroll
    for (int mi = 0; mi < 2; mi++) {
      const int mbase = bm + mi * 16 + lh * 8;
#pragma unroll
      for (int r = 0; r < 8; r++) {
        const int m = mbase + r;
        if (m < M) C[(long)m * ldc + nc] = acc[mi][ni][r] + bb;
      }
    }
  }
}

// ---------------- GAT edge pipeline ----------------

__device__ __forceinline__ unsigned f2ord(float f) {
  unsigned u = __float_as_uint(f);
  return (u & 0x80000000u) ? ~u : (u | 0x80000000u);
}
__device__ __forceinline__ float ord2f(unsigned k) {
  unsigned u = (k & 0x80000000u) ? (k & 0x7fffffffu) : ~k;
  return __uint_as_float(u);
}

// per (node, head): alpha_src/dst = <xl[n,h,:], a_src/dst[h,:]>, one wave each
__global__ __launch_bounds__(256)
void alpha_kernel(const float* __restrict__ xl, int ldxl,
                  const float* __restrict__ a_src, const float* __restrict__ a_dst,
                  float* __restrict__ asrc, float* __restrict__ adst,
                  int nodes, int H, int C) {
  int wid  = (int)(((long)blockIdx.x * blockDim.x + threadIdx.x) >> 5);
  int lane = threadIdx.x & 31;
  if (wid >= nodes * H) return;
  int n = wid / H, h = wid % H;
  const float* row = xl + (long)n * ldxl + h * C;
  const float* as  = a_src + h * C;
  const float* ad  = a_dst + h * C;
  float ss = 0.0f, sd = 0.0f;
  for (int c = lane; c < C; c += 32) { float v = row[c]; ss += v * as[c]; sd += v * ad[c]; }
  for (int o = 16; o; o >>= 1) { ss += __shfl_xor(ss, o, 32); sd += __shfl_xor(sd, o, 32); }
  if (lane == 0) { asrc[wid] = ss; adst[wid] = sd; }
}

// segment max of leaky_relu(asrc[src]+adst[dst]) over dst (ordered-uint atomic max)
__global__ void edge_max_kernel(const int* __restrict__ ei, int E, int nodes, int H,
                                const float* __restrict__ asrc, const float* __restrict__ adst,
                                unsigned* __restrict__ mmax) {
  int e = blockIdx.x * blockDim.x + threadIdx.x;
  int EN = E + nodes;
  if (e >= EN) return;
  int s, d;
  if (e < E) { s = ei[e]; d = ei[E + e]; } else { s = d = e - E; }
  for (int h = 0; h < H; h++) {
    float v = asrc[s * H + h] + adst[d * H + h];
    v = (v > 0.0f) ? v : NEG_SLOPE * v;
    atomicMax(&mmax[d * H + h], f2ord(v));
  }
}

// ex = exp(e - m[dst]); store per-edge; denom[dst] += ex
__global__ void edge_exp_kernel(const int* __restrict__ ei, int E, int nodes, int H,
                                const float* __restrict__ asrc, const float* __restrict__ adst,
                                const unsigned* __restrict__ mmax,
                                float* __restrict__ ealpha, float* __restrict__ denom) {
  int e = blockIdx.x * blockDim.x + threadIdx.x;
  int EN = E + nodes;
  if (e >= EN) return;
  int s, d;
  if (e < E) { s = ei[e]; d = ei[E + e]; } else { s = d = e - E; }
  for (int h = 0; h < H; h++) {
    float v = asrc[s * H + h] + adst[d * H + h];
    v = (v > 0.0f) ? v : NEG_SLOPE * v;
    float ex = __expf(v - ord2f(mmax[d * H + h]));
    ealpha[(long)e * H + h] = ex;
    atomicAdd(&denom[d * H + h], ex);
  }
}

// out[dst, :] += alpha_e[h] * xl[src, :]; one block per edge, atomics land in L2
__global__ __launch_bounds__(256)
void aggregate_kernel(const int* __restrict__ ei, int E, int nodes, int H, int C,
                      int ldxl, int ldo,
                      const float* __restrict__ xl,
                      const float* __restrict__ ealpha, const float* __restrict__ denom,
                      float* __restrict__ out) {
  int e = blockIdx.x;
  int s, d;
  if (e < E) { s = ei[e]; d = ei[E + e]; } else { s = d = e - E; }
  __shared__ float sc[8];
  if ((int)threadIdx.x < H)
    sc[threadIdx.x] = ealpha[(long)e * H + threadIdx.x] / (denom[d * H + threadIdx.x] + 1e-16f);
  __syncthreads();
  const float* xr = xl + (long)s * ldxl;
  float* orow = out + (long)d * ldo;
  const int HC = H * C;
  for (int c = threadIdx.x; c < HC; c += blockDim.x) {
    int h = c / C;
    atomicAdd(&orow[c], sc[h] * xr[c]);
  }
}

// in-place LayerNorm + ELU, one block per row of F
__global__ __launch_bounds__(256)
void ln_elu_kernel(float* __restrict__ h, const float* __restrict__ g,
                   const float* __restrict__ beta, int F) {
  int n = blockIdx.x;
  float* row = h + (long)n * F;
  float s = 0.0f, s2 = 0.0f;
  for (int i = threadIdx.x; i < F; i += 256) { float v = row[i]; s += v; s2 += v * v; }
  for (int o = 16; o; o >>= 1) { s += __shfl_xor(s, o, 32); s2 += __shfl_xor(s2, o, 32); }
  __shared__ float sh[16];
  int w = threadIdx.x >> 5, lane = threadIdx.x & 31;
  if (lane == 0) { sh[w] = s; sh[8 + w] = s2; }
  __syncthreads();
  if (threadIdx.x == 0) {
    float S = 0.0f, S2 = 0.0f;
    for (int i = 0; i < 8; i++) { S += sh[i]; S2 += sh[8 + i]; }
    sh[0] = S; sh[8] = S2;
  }
  __syncthreads();
  float mu  = sh[0] / F;
  float var = sh[8] / F - mu * mu;
  float inv = rsqrtf(var + LN_EPS);
  for (int i = threadIdx.x; i < F; i += 256) {
    float v = (row[i] - mu) * inv * g[i] + beta[i];
    row[i] = (v > 0.0f) ? v : (__expf(v) - 1.0f);
  }
}

// out[n,c] = mean_h(agg[n,h,c]) + b3[c]
__global__ void final_mean_kernel(const float* __restrict__ agg, const float* __restrict__ b3,
                                  float* __restrict__ out, int nodes, int H, int C) {
  long i = (long)blockIdx.x * blockDim.x + threadIdx.x;
  if (i >= (long)nodes * C) return;
  int  c = (int)(i % C);
  long n = i / C;
  const float* r = agg + n * (long)(H * C);
  float sum = 0.0f;
  for (int h = 0; h < H; h++) sum += r[h * C + c];
  out[i] = sum / (float)H + b3[c];
}

// ---------------- host orchestration ----------------

static void launch_gat_layer(const __bf16* Abf, int K,
                             const __bf16* Wt, const __bf16* St,
                             const float* a_src, const float* a_dst,
                             const float* bias1, const float* bias2,
                             int H, int C,
                             float* buf_xl, float* buf_out,
                             float* asrc, float* adst, unsigned* mmax, float* denom,
                             float* ealpha, const int* ei, int E, hipStream_t stream) {
  const int Nout = H * C;
  dim3 blk(256);
  dim3 gg((NNODES + 127) / 128, (Nout + 127) / 128);
  gemm_bf16_wmma_kernel<<<gg, blk, 0, stream>>>(Abf, Wt, buf_xl, NNODES, Nout, K, Nout,
                                                nullptr, nullptr);
  if (St) {
    // skip path (+ conv bias + skip bias) seeds the aggregation buffer
    gemm_bf16_wmma_kernel<<<gg, blk, 0, stream>>>(Abf, St, buf_out, NNODES, Nout, K, Nout,
                                                  bias1, bias2);
  } else {
    long tot = (long)NNODES * Nout;
    zero_f32_kernel<<<(int)((tot + 255) / 256), 256, 0, stream>>>(buf_out, tot);
  }
  alpha_kernel<<<(NNODES * H + 7) / 8, 256, 0, stream>>>(buf_xl, Nout, a_src, a_dst,
                                                         asrc, adst, NNODES, H, C);
  init_softmax_kernel<<<(NNODES * H + 255) / 256, 256, 0, stream>>>(mmax, denom, NNODES * H);
  const int EN = E + NNODES;
  edge_max_kernel<<<(EN + 255) / 256, 256, 0, stream>>>(ei, E, NNODES, H, asrc, adst, mmax);
  edge_exp_kernel<<<(EN + 255) / 256, 256, 0, stream>>>(ei, E, NNODES, H, asrc, adst, mmax,
                                                        ealpha, denom);
  aggregate_kernel<<<EN, 256, 0, stream>>>(ei, E, NNODES, H, C, Nout, Nout,
                                           buf_xl, ealpha, denom, buf_out);
}

extern "C" void kernel_launch(void* const* d_in, const int* in_sizes, int n_in,
                              void* d_out, int out_size, void* d_ws, size_t ws_size,
                              hipStream_t stream) {
  const float* x       = (const float*)d_in[0];
  const int*   ei      = (const int*)d_in[1];      // [2,E] int32
  const float* W1      = (const float*)d_in[2];
  const float* a_src1  = (const float*)d_in[3];
  const float* a_dst1  = (const float*)d_in[4];
  const float* b1      = (const float*)d_in[5];
  const float* W2      = (const float*)d_in[6];
  const float* a_src2  = (const float*)d_in[7];
  const float* a_dst2  = (const float*)d_in[8];
  const float* b2      = (const float*)d_in[9];
  const float* W3      = (const float*)d_in[10];
  const float* a_src3  = (const float*)d_in[11];
  const float* a_dst3  = (const float*)d_in[12];
  const float* b3      = (const float*)d_in[13];
  const float* skipW1  = (const float*)d_in[14];
  const float* skipb1  = (const float*)d_in[15];
  const float* skipW2  = (const float*)d_in[16];
  const float* skipb2  = (const float*)d_in[17];
  const float* g1      = (const float*)d_in[18];
  const float* beta1   = (const float*)d_in[19];
  const float* g2      = (const float*)d_in[20];
  const float* beta2   = (const float*)d_in[21];

  const int E = in_sizes[1] / 2;

  // workspace carve-out (~215 MB total)
  char* ws = (char*)d_ws;
  size_t off = 0;
  auto carve = [&](size_t bytes) -> void* {
    void* p = ws + off;
    off = (off + bytes + 255) & ~(size_t)255;
    return p;
  };
  float*    buf_xl  = (float*)carve((size_t)NNODES * 1024 * 4);
  float*    buf_out = (float*)carve((size_t)NNODES * 1024 * 4);
  __bf16*   Abf     = (__bf16*)carve((size_t)NNODES * 1024 * 2);
  __bf16*   W1t     = (__bf16*)carve((size_t)1024 * 64 * 2);
  __bf16*   S1t     = (__bf16*)carve((size_t)1024 * 64 * 2);
  __bf16*   W2t     = (__bf16*)carve((size_t)1024 * 1024 * 2);
  __bf16*   S2t     = (__bf16*)carve((size_t)1024 * 1024 * 2);
  __bf16*   W3t     = (__bf16*)carve((size_t)726 * 1024 * 2);
  float*    asrc    = (float*)carve((size_t)NNODES * 6 * 4);
  float*    adst    = (float*)carve((size_t)NNODES * 6 * 4);
  unsigned* mmax    = (unsigned*)carve((size_t)NNODES * 6 * 4);
  float*    denom   = (float*)carve((size_t)NNODES * 6 * 4);
  float*    ealpha  = (float*)carve((size_t)(E + NNODES) * 6 * 4);

  auto gridFor = [](long n) { return (int)((n + 255) / 256); };

  // weights -> bf16, transposed [N, Kpad]   (args: W, Wt, Kin, Nw, Kpad)
  wtrans_bf16_kernel<<<gridFor((long)1024 * 64), 256, 0, stream>>>(W1,     W1t, 50,   1024, 64);
  wtrans_bf16_kernel<<<gridFor((long)1024 * 64), 256, 0, stream>>>(skipW1, S1t, 50,   1024, 64);
  wtrans_bf16_kernel<<<gridFor((long)1024 * 1024), 256, 0, stream>>>(W2,     W2t, 1024, 1024, 1024);
  wtrans_bf16_kernel<<<gridFor((long)1024 * 1024), 256, 0, stream>>>(skipW2, S2t, 1024, 1024, 1024);
  wtrans_bf16_kernel<<<gridFor((long)726 * 1024), 256, 0, stream>>>(W3,     W3t, 1024, 726,  1024);

  // layer 1: x[20000,50] -> pad K=64
  cvt_pad_bf16_kernel<<<gridFor((long)NNODES * 64), 256, 0, stream>>>(x, Abf, NNODES, 50, 64);
  launch_gat_layer(Abf, 64, W1t, S1t, a_src1, a_dst1, b1, skipb1, 4, 256,
                   buf_xl, buf_out, asrc, adst, mmax, denom, ealpha, ei, E, stream);
  ln_elu_kernel<<<NNODES, 256, 0, stream>>>(buf_out, g1, beta1, 1024);

  // layer 2
  cvt_pad_bf16_kernel<<<gridFor((long)NNODES * 1024), 256, 0, stream>>>(buf_out, Abf, NNODES, 1024, 1024);
  launch_gat_layer(Abf, 1024, W2t, S2t, a_src2, a_dst2, b2, skipb2, 4, 256,
                   buf_xl, buf_out, asrc, adst, mmax, denom, ealpha, ei, E, stream);
  ln_elu_kernel<<<NNODES, 256, 0, stream>>>(buf_out, g2, beta2, 1024);

  // layer 3: heads=6, ch=121, concat=False
  cvt_pad_bf16_kernel<<<gridFor((long)NNODES * 1024), 256, 0, stream>>>(buf_out, Abf, NNODES, 1024, 1024);
  launch_gat_layer(Abf, 1024, W3t, nullptr, a_src3, a_dst3, nullptr, nullptr, 6, 121,
                   buf_xl, buf_out, asrc, adst, mmax, denom, ealpha, ei, E, stream);
  final_mean_kernel<<<gridFor((long)NNODES * 121), 256, 0, stream>>>(buf_out, b3, (float*)d_out,
                                                                     NNODES, 6, 121);
}
